// Pooling_block_6493990551773
// MI455X (gfx1250) — matrix operand order repaired and checked by reading, not compiled, not run on hardware
//
#include <hip/hip_runtime.h>

typedef __attribute__((ext_vector_type(16))) _Float16 v16h;
typedef __attribute__((ext_vector_type(8)))  float    v8f;

#define B_    256
#define HW_   196
#define C_    384
#define NX_   (HW_ + 1)   // 197 rows in x
#define START_ 147
#define OUTR_  50
#define TPB_   256
#define NTILE_ ((HW_ + 15) / 16)   // 13
#define C4_   (C_ / 4)             // 96 float4 columns

__global__ __launch_bounds__(TPB_)
void pool_fused_kernel(const float* __restrict__ x,
                       const float* __restrict__ ea,
                       float* __restrict__ out) {
  __shared__ float4   s_part[2][C4_];  // phase-1 partial sums (3 KB)
  __shared__ _Float16 s_wh[C_];        // sigmoid channel importance (f16, WMMA B)
  __shared__ float    s_key[TPB_];     // scores padded to 256 with +inf
  __shared__ int      s_idx[TPB_];     // permutation

  const int b   = blockIdx.x;
  const int tid = threadIdx.x;

  // ---- Phase 1a: columnwise partial sums of ea[b,:,:], float4 streams -------
  // 192 threads: col = tid % 96, row-half = tid / 96 (98 rows each).
  // Per-wave access is 32 consecutive float4 = 512B contiguous -> coalesced.
  {
    const int col = tid % C4_;
    const int rg  = tid / C4_;
    if (rg < 2) {
      const float4* p = (const float4*)(ea + (size_t)b * HW_ * C_) + col;
      const int n0 = rg * (HW_ / 2);
      float4 s = make_float4(0.f, 0.f, 0.f, 0.f);
      for (int n = n0; n < n0 + HW_ / 2; ++n) {
        const float4 v = p[(size_t)n * C4_];
        s.x += v.x; s.y += v.y; s.z += v.z; s.w += v.w;
      }
      s_part[rg][col] = s;
    }
  }
  __syncthreads();

  // ---- Phase 1b: combine halves, sigmoid(mean), store f16 weights -----------
  if (tid < C4_) {
    const float4 a = s_part[0][tid];
    const float4 c = s_part[1][tid];
    const float inv = 1.0f / (float)HW_;
    s_wh[tid * 4 + 0] = (_Float16)(1.0f / (1.0f + __expf(-(a.x + c.x) * inv)));
    s_wh[tid * 4 + 1] = (_Float16)(1.0f / (1.0f + __expf(-(a.y + c.y) * inv)));
    s_wh[tid * 4 + 2] = (_Float16)(1.0f / (1.0f + __expf(-(a.z + c.z) * inv)));
    s_wh[tid * 4 + 3] = (_Float16)(1.0f / (1.0f + __expf(-(a.w + c.w) * inv)));
  }
  __syncthreads();

  // ---- Phase 2: scores via V_WMMA_F32_16X16X32_F16 --------------------------
  // C(16x16) = A(16 node rows x 32 ch) * B(w chunk replicated across columns).
  // Row index is CLAMPED (not branched); duplicate scores land in s_key slots
  // 196..207 and are overwritten with +inf after the barrier below.
  {
    const int wave = tid >> 5;
    const int lane = tid & 31;
    const int mloc = lane & 15;
    const int hi   = lane >> 4;
    for (int t = wave; t < NTILE_; t += TPB_ / 32) {   // wave-uniform trip
      const int m0   = t * 16;
      const int row  = m0 + mloc;
      const int rowc = row < HW_ ? row : HW_ - 1;      // clamp, stay in-bounds
      const float* rowp = x + ((size_t)b * NX_ + rowc + 1) * C_;
      v8f acc = {};
      for (int kc = 0; kc < C_; kc += 32) {
        const int kbA = kc + hi * 8;    // A: lane<16 -> K {0..7,16..23}; >=16 -> {8..15,24..31}
        const int kbB = kc + hi * 8;    // B: lane<16 -> K {0..7,16..23}; >=16 -> {8..15,24..31}
        v16h av, bv;
        #pragma unroll
        for (int e = 0; e < 16; ++e) bv[e] = s_wh[kbB + e + (e >= 8 ? 8 : 0)];
        const float4 u0 = *(const float4*)(rowp + kbA);
        const float4 u1 = *(const float4*)(rowp + kbA + 4);
        const float4 u2 = *(const float4*)(rowp + kbA + 16);
        const float4 u3 = *(const float4*)(rowp + kbA + 20);
        av[0]  = (_Float16)u0.x; av[1]  = (_Float16)u0.y;
        av[2]  = (_Float16)u0.z; av[3]  = (_Float16)u0.w;
        av[4]  = (_Float16)u1.x; av[5]  = (_Float16)u1.y;
        av[6]  = (_Float16)u1.z; av[7]  = (_Float16)u1.w;
        av[8]  = (_Float16)u2.x; av[9]  = (_Float16)u2.y;
        av[10] = (_Float16)u2.z; av[11] = (_Float16)u2.w;
        av[12] = (_Float16)u3.x; av[13] = (_Float16)u3.y;
        av[14] = (_Float16)u3.z; av[15] = (_Float16)u3.w;
        acc = __builtin_amdgcn_wmma_f32_16x16x32_f16(
                  /*neg_a=*/false, av, /*neg_b=*/false, bv,
                  /*c_mod=*/(short)0, acc, /*reuse_a=*/false, /*reuse_b=*/false);
      }
      // C layout: VGPR r, lanes 0-15 -> M=r; lanes 16-31 -> M=8+r.
      // Lane 0 holds scores m0..m0+7, lane 16 holds m0+8..m0+15 (col N=0).
      // Unconditional 2x ds_store_b128 under a single lane predicate.
      if ((lane & 15) == 0) {
        float4* dst = (float4*)&s_key[m0 + hi * 8];
        dst[0] = make_float4(acc[0], acc[1], acc[2], acc[3]);
        dst[1] = make_float4(acc[4], acc[5], acc[6], acc[7]);
      }
    }
  }
  __syncthreads();                        // tile-12 garbage writes complete
  s_idx[tid] = tid;
  if (tid >= HW_) s_key[tid] = 1.0e30f;   // pads sort past position 195
  __syncthreads();

  // ---- Phase 3: ascending bitonic sort of (key, idx) ------------------------
  for (int k = 2; k <= TPB_; k <<= 1) {
    for (int j = k >> 1; j > 0; j >>= 1) {
      const int i   = tid;
      const int ixj = i ^ j;
      if (ixj > i) {
        const float ki = s_key[i], kj = s_key[ixj];
        const bool asc = ((i & k) == 0);
        if (asc ? (ki > kj) : (ki < kj)) {
          const int ti = s_idx[i];
          s_key[i] = kj;  s_key[ixj] = ki;
          s_idx[i] = s_idx[ixj]; s_idx[ixj] = ti;
        }
      }
      __syncthreads();
    }
  }

  // ---- Phase 4: gather top-49 rows + CLS row, float4 copies -----------------
  for (int q = tid; q < OUTR_ * C4_; q += TPB_) {
    const int r   = q / C4_;
    const int col = q - r * C4_;
    const int srcRow = (r == 0) ? 0 : (s_idx[START_ + r - 1] + 1);
    const float4* src = (const float4*)(x + ((size_t)b * NX_ + srcRow) * C_) + col;
    float4*       dst = (float4*)(out + ((size_t)b * OUTR_ + r) * C_) + col;
    *dst = *src;
  }
}

extern "C" void kernel_launch(void* const* d_in, const int* in_sizes, int n_in,
                              void* d_out, int out_size, void* d_ws, size_t ws_size,
                              hipStream_t stream) {
  (void)in_sizes; (void)n_in; (void)out_size; (void)d_ws; (void)ws_size;
  const float* x   = (const float*)d_in[0];
  const float* ea  = (const float*)d_in[1];
  float*       out = (float*)d_out;
  pool_fused_kernel<<<dim3(B_), dim3(TPB_), 0, stream>>>(x, ea, out);
}
